// InterpretableMultiHeadAttention_78632261256105
// MI455X (gfx1250) — compile-verified, matrix operand order checked by reading, not run on hardware
//
#include <hip/hip_runtime.h>
#include <hip/hip_bf16.h>

// ---------------------------------------------------------------------------
// InterpretableMultiHeadAttention for MI455X (gfx1250, wave32, WMMA).
// - All GEMMs through v_wmma_f32_16x16x32_bf16, f32 accumulate.
// - Q/K projections emit bf16 directly so the attention kernel streams raw
//   bf16 tiles into LDS (async-to-LDS when the builtin exists), no VALU cvt.
// ---------------------------------------------------------------------------

typedef __attribute__((ext_vector_type(16))) __bf16 v16bf;
typedef __attribute__((ext_vector_type(8)))  float  v8f;
typedef __attribute__((ext_vector_type(4)))  int    v4i;

#define SEQ_LEN   2048
#define BATCH     4
#define EMBED     1024
#define NHEADS    8
#define HDIM      128
#define ROWS      8192          // SEQ_LEN * BATCH
#define NEG_INF   (-3.0e38f)

__device__ __forceinline__ unsigned f2bfu(float f) {
  unsigned u = __builtin_bit_cast(unsigned, f);
  return (u + 0x7FFFu + ((u >> 16) & 1u)) >> 16;   // round-nearest-even bits
}
__device__ __forceinline__ unsigned pk2bf(float a, float b) {
  return f2bfu(a) | (f2bfu(b) << 16);
}

// ---- async global->LDS (CDNA5) with compile-safe fallback -----------------
#if defined(__HIP_DEVICE_COMPILE__) && __has_builtin(__builtin_amdgcn_global_load_async_to_lds_b128)
#define HAVE_ASYNC_LDS 1
#else
#define HAVE_ASYNC_LDS 0
#endif

__device__ __forceinline__ void copy16_g2lds(const void* g, void* l) {
#if HAVE_ASYNC_LDS
  __builtin_amdgcn_global_load_async_to_lds_b128(
      (__attribute__((address_space(1))) v4i*)(v4i*)(void*)g,
      (__attribute__((address_space(3))) v4i*)(v4i*)l, 0, 0);
#else
  *(uint4*)l = *(const uint4*)g;
#endif
}
__device__ __forceinline__ void wait_async_lds() {
#if HAVE_ASYNC_LDS
#if __has_builtin(__builtin_amdgcn_s_wait_asynccnt)
  __builtin_amdgcn_s_wait_asynccnt(0);
#else
  asm volatile("s_wait_asynccnt 0x0" ::: "memory");
#endif
#endif
}

// ---------------------------------------------------------------------------
// Generic WMMA GEMM:  val = alpha * (A[M,K] @ B[K,N] + bias[N])
//   Cf  (f32)  written if non-null
//   Cbf (bf16) written if non-null
// Block tile 128x128, KT=64, 8 waves (2x4), wave tile 64x32 (4x2 WMMA tiles).
// ---------------------------------------------------------------------------
#define KT 64
#define APAD 80   // 64+16 bf16 -> 160B row stride (32B aligned)

__global__ __launch_bounds__(256) void wmma_gemm_kernel(
    const float* __restrict__ A, int lda,
    const float* __restrict__ B, int ldb,
    const float* __restrict__ bias,
    float* __restrict__ Cf,
    unsigned short* __restrict__ Cbf, int ldc,
    int M, int N, int K, float alpha)
{
  __shared__ __bf16 As[128][APAD];   // [m][k]
  __shared__ __bf16 Bs[128][APAD];   // [n][k]  (B staged transposed)

  const int tid  = threadIdx.x;
  const int wave = tid >> 5;
  const int lane = tid & 31;
  const int lr   = lane & 15;
  const int lh   = lane >> 4;
  const int wm   = wave >> 2;        // 0..1
  const int wn   = wave & 3;         // 0..3
  const int m0   = blockIdx.y * 128;
  const int n0   = blockIdx.x * 128;

  v8f acc[4][2];
#pragma unroll
  for (int mi = 0; mi < 4; ++mi)
#pragma unroll
    for (int ni = 0; ni < 2; ++ni)
      acc[mi][ni] = (v8f){0.f,0.f,0.f,0.f,0.f,0.f,0.f,0.f};

  for (int k0 = 0; k0 < K; k0 += KT) {
    // Stage A tile 128x64: float4 loads -> packed bf16x2 LDS stores
#pragma unroll
    for (int i = 0; i < 8; ++i) {
      int idx = tid + i * 256;              // 0..2047 float4 slots
      int r = idx >> 4, c = (idx & 15) * 4;
      const float4 v4 = *(const float4*)(A + (size_t)(m0 + r) * lda + k0 + c);
      unsigned* dst = (unsigned*)&As[r][c];
      dst[0] = pk2bf(v4.x, v4.y);
      dst[1] = pk2bf(v4.z, v4.w);
    }
    // Stage B tile 64x128 transposed -> Bs[n][k], packed k-pairs
#pragma unroll
    for (int i = 0; i < 16; ++i) {
      int idx = tid + i * 256;              // 0..4095 k-pair slots
      int k2 = (idx >> 7) * 2, nn = idx & 127;
      const float f0 = B[(size_t)(k0 + k2) * ldb + n0 + nn];
      const float f1 = B[(size_t)(k0 + k2 + 1) * ldb + n0 + nn];
      *(unsigned*)&Bs[nn][k2] = pk2bf(f0, f1);
    }
    if (k0 + KT < K) {                      // hint next tiles toward L2/L0
      __builtin_prefetch(A + (size_t)(m0 + (tid >> 1)) * lda + k0 + KT, 0, 0);
      __builtin_prefetch(B + (size_t)(k0 + KT + (tid >> 2)) * ldb + n0 + (tid & 3) * 32, 0, 0);
    }
    __syncthreads();

#pragma unroll
    for (int ks = 0; ks < 2; ++ks) {
      const int kb = ks * 32 + lh * 16;
      v16bf afr[4], bfr[2];
#pragma unroll
      for (int mi = 0; mi < 4; ++mi) {
        const __bf16* p = &As[wm * 64 + mi * 16 + lr][kb];
#pragma unroll
        for (int j = 0; j < 16; ++j) afr[mi][j] = p[j];
      }
#pragma unroll
      for (int ni = 0; ni < 2; ++ni) {
        const __bf16* p = &Bs[wn * 32 + ni * 16 + lr][kb];
#pragma unroll
        for (int j = 0; j < 16; ++j) bfr[ni][j] = p[j];
      }
#pragma unroll
      for (int mi = 0; mi < 4; ++mi)
#pragma unroll
        for (int ni = 0; ni < 2; ++ni)
          acc[mi][ni] = __builtin_amdgcn_wmma_f32_16x16x32_bf16(
              false, afr[mi], false, bfr[ni], (short)0, acc[mi][ni], false, false);
    }
    __syncthreads();
  }

  // Epilogue (ISA C layout: VGPR j -> row j / j+8 for lane halves, col = lr)
#pragma unroll
  for (int mi = 0; mi < 4; ++mi)
#pragma unroll
    for (int ni = 0; ni < 2; ++ni) {
      const int col = n0 + wn * 32 + ni * 16 + lr;
      const float bb = bias ? bias[col] : 0.f;
#pragma unroll
      for (int j = 0; j < 8; ++j) {
        const int row = m0 + wm * 64 + mi * 16 + j + lh * 8;
        const float val = alpha * (acc[mi][ni][j] + bb);
        if (Cf)  Cf[(size_t)row * ldc + col] = val;
        if (Cbf) Cbf[(size_t)row * ldc + col] = (unsigned short)f2bfu(val);
      }
    }
}

// ---------------------------------------------------------------------------
// Causal scores + per-head softmax + head-mean weights.
// One workgroup = 16 query rows of one batch. qp/kp are bf16 (pre-scaled Q).
// ---------------------------------------------------------------------------
#define QPAD 144   // 128+16 bf16 -> 288B stride (16B aligned rows)

__global__ __launch_bounds__(256) void attn_softmax_kernel(
    const unsigned short* __restrict__ qpb,   // (l,n,EMBED) bf16 bits
    const unsigned short* __restrict__ kpb,   // (m,n,EMBED) bf16 bits
    float* __restrict__ wmean)                // (n, SEQ_LEN, SEQ_LEN)
{
  __shared__ __bf16 Qs[16][QPAD];
  __shared__ __bf16 Ks[128][QPAD];
  __shared__ float  Ss[16][128];
  __shared__ float  meanS[16][128];
  __shared__ float  Mrow[NHEADS][16];
  __shared__ float  Lrow[NHEADS][16];

  const int tid  = threadIdx.x;
  const int wave = tid >> 5;
  const int lane = tid & 31;
  const int lr   = lane & 15;
  const int lh   = lane >> 4;
  const int l0   = blockIdx.x * 16;
  const int n    = blockIdx.y;
  const int ntiles = (l0 + 15) / 128 + 1;     // causal column tiles

  // ---------------- Phase 1: per-head running (max, sumexp) ----------------
  for (int h = 0; h < NHEADS; ++h) {
    {   // Q block 16x128 bf16: one 16B chunk per thread
      int r = tid >> 4, ch = (tid & 15) * 8;
      copy16_g2lds(qpb + ((size_t)(l0 + r) * BATCH + n) * EMBED + h * HDIM + ch,
                   &Qs[r][ch]);
    }
    if (tid < 16) { Mrow[h][tid] = NEG_INF; Lrow[h][tid] = 0.f; }

    for (int ct = 0; ct < ntiles; ++ct) {
#pragma unroll
      for (int i = 0; i < 8; ++i) {           // K tile 128x128 bf16
        int idx = tid + i * 256;
        int r = idx >> 4, ch = (idx & 15) * 8;
        copy16_g2lds(kpb + ((size_t)(ct * 128 + r) * BATCH + n) * EMBED + h * HDIM + ch,
                     &Ks[r][ch]);
      }
      wait_async_lds();
      __syncthreads();

      v8f acc = (v8f){0.f,0.f,0.f,0.f,0.f,0.f,0.f,0.f};
#pragma unroll
      for (int ks = 0; ks < 4; ++ks) {        // e = 128 contraction
        const int kb = ks * 32 + lh * 16;
        v16bf af, bf;
        const __bf16* pa = &Qs[lr][kb];
        const __bf16* pb = &Ks[wave * 16 + lr][kb];
#pragma unroll
        for (int j = 0; j < 16; ++j) { af[j] = pa[j]; bf[j] = pb[j]; }
        acc = __builtin_amdgcn_wmma_f32_16x16x32_bf16(
            false, af, false, bf, (short)0, acc, false, false);
      }

      const int scol = ct * 128 + wave * 16 + lr;
#pragma unroll
      for (int j = 0; j < 8; ++j) {
        const int srow = l0 + j + lh * 8;
        float v = acc[j];
        if (scol > srow) v = NEG_INF;         // causal mask
        Ss[j + lh * 8][wave * 16 + lr] = v;
      }
      __syncthreads();

      if (tid < 16) {                          // per-row online merge
        float tm = NEG_INF;
        for (int c = 0; c < 128; ++c) tm = fmaxf(tm, Ss[tid][c]);
        float tl = 0.f;
        if (tm > -1.0e37f)
          for (int c = 0; c < 128; ++c) tl += __expf(Ss[tid][c] - tm);
        const float mo = Mrow[h][tid], lo = Lrow[h][tid];
        const float mn = fmaxf(mo, tm);
        Lrow[h][tid] = lo * __expf(mo - mn) + tl * __expf(tm - mn);
        Mrow[h][tid] = mn;
      }
      __syncthreads();
    }
  }

  // ---------------- Phase 2: probabilities + head mean ----------------
  const size_t wbase = (size_t)n * SEQ_LEN * SEQ_LEN;
  for (int ct = 0; ct < ntiles; ++ct) {
#pragma unroll
    for (int i = 0; i < 8; ++i) {             // zero mean tile
      int idx = tid + i * 256;
      meanS[idx >> 7][idx & 127] = 0.f;
    }
    __syncthreads();

    for (int h = 0; h < NHEADS; ++h) {
      {
        int r = tid >> 4, ch = (tid & 15) * 8;
        copy16_g2lds(qpb + ((size_t)(l0 + r) * BATCH + n) * EMBED + h * HDIM + ch,
                     &Qs[r][ch]);
      }
#pragma unroll
      for (int i = 0; i < 8; ++i) {
        int idx = tid + i * 256;
        int r = idx >> 4, ch = (idx & 15) * 8;
        copy16_g2lds(kpb + ((size_t)(ct * 128 + r) * BATCH + n) * EMBED + h * HDIM + ch,
                     &Ks[r][ch]);
      }
      wait_async_lds();
      __syncthreads();

      v8f acc = (v8f){0.f,0.f,0.f,0.f,0.f,0.f,0.f,0.f};
#pragma unroll
      for (int ks = 0; ks < 4; ++ks) {
        const int kb = ks * 32 + lh * 16;
        v16bf af, bf;
        const __bf16* pa = &Qs[lr][kb];
        const __bf16* pb = &Ks[wave * 16 + lr][kb];
#pragma unroll
        for (int j = 0; j < 16; ++j) { af[j] = pa[j]; bf[j] = pb[j]; }
        acc = __builtin_amdgcn_wmma_f32_16x16x32_bf16(
            false, af, false, bf, (short)0, acc, false, false);
      }

      const int scol = ct * 128 + wave * 16 + lr;
#pragma unroll
      for (int j = 0; j < 8; ++j) {
        const int rloc = j + lh * 8;
        const int srow = l0 + rloc;
        float w = 0.f;
        if (scol <= srow)
          w = __expf(acc[j] - Mrow[h][rloc]) / Lrow[h][rloc] * (1.0f / NHEADS);
        meanS[rloc][wave * 16 + lr] += w;      // unique addr per lane/j
      }
      __syncthreads();
    }

#pragma unroll
    for (int i = 0; i < 8; ++i) {              // write head-mean tile
      int idx = tid + i * 256;
      int r = idx >> 7, c = idx & 127;
      wmean[wbase + (size_t)(l0 + r) * SEQ_LEN + ct * 128 + c] = meanS[r][c];
    }
    __syncthreads();
  }

  // Zero the fully-masked remainder of each row (d_out is poisoned).
  for (int c = ntiles * 128 + tid; c < SEQ_LEN; c += 256)
    for (int r = 0; r < 16; ++r)
      wmean[wbase + (size_t)(l0 + r) * SEQ_LEN + c] = 0.f;
}

// ---------------------------------------------------------------------------
extern "C" void kernel_launch(void* const* d_in, const int* in_sizes, int n_in,
                              void* d_out, int out_size, void* d_ws, size_t ws_size,
                              hipStream_t stream) {
  (void)in_sizes; (void)n_in; (void)out_size; (void)ws_size;
  const float* q  = (const float*)d_in[0];
  const float* k  = (const float*)d_in[1];
  const float* v  = (const float*)d_in[2];
  const float* Wq = (const float*)d_in[3];
  const float* bq = (const float*)d_in[4];
  const float* Wk = (const float*)d_in[5];
  const float* bk = (const float*)d_in[6];
  const float* Wv = (const float*)d_in[7];
  const float* bv = (const float*)d_in[8];
  const float* Wo = (const float*)d_in[9];
  const float* bo = (const float*)d_in[10];

  float* out   = (float*)d_out;                               // (2048,4,1024)
  float* wmean = out + (size_t)SEQ_LEN * BATCH * EMBED;       // (4,2048,2048)

  // Workspace: qp/kp as bf16 (16 MB each), vp f32 (32 MB), attn f32 (32 MB)
  unsigned short* qpb = (unsigned short*)d_ws;
  unsigned short* kpb = qpb + (size_t)ROWS * EMBED;
  float* vp   = (float*)(kpb + (size_t)ROWS * EMBED);
  float* attn = vp + (size_t)ROWS * EMBED;

  const dim3 blk(256);
  const dim3 gproj(EMBED / 128, ROWS / 128);                  // (8, 64)
  const float scaling = 0.08838834764831845f;                 // HEAD_DIM^-0.5

  // Projections: Q/K emit bf16 for the attention kernel, V stays f32
  wmma_gemm_kernel<<<gproj, blk, 0, stream>>>(q, EMBED, Wq, EMBED, bq,
                                              nullptr, qpb, EMBED,
                                              ROWS, EMBED, EMBED, scaling);
  wmma_gemm_kernel<<<gproj, blk, 0, stream>>>(k, EMBED, Wk, EMBED, bk,
                                              nullptr, kpb, EMBED,
                                              ROWS, EMBED, EMBED, 1.0f);
  wmma_gemm_kernel<<<gproj, blk, 0, stream>>>(v, EMBED, Wv, EMBED, bv,
                                              vp, nullptr, EMBED,
                                              ROWS, EMBED, EMBED, 1.0f);

  // Scores + softmax + head-averaged weights (second output)
  attn_softmax_kernel<<<dim3(SEQ_LEN / 16, BATCH), blk, 0, stream>>>(qpb, kpb, wmean);

  // attn[l,n,:] = w_mean[n] @ vp[:,n,:]   (one GEMM per batch)
  const dim3 gattn(EMBED / 128, SEQ_LEN / 128);               // (8, 16)
  for (int nn = 0; nn < BATCH; ++nn)
    wmma_gemm_kernel<<<gattn, blk, 0, stream>>>(
        wmean + (size_t)nn * SEQ_LEN * SEQ_LEN, SEQ_LEN,
        vp + nn * EMBED, BATCH * EMBED, nullptr,
        attn + nn * EMBED, nullptr, BATCH * EMBED,
        SEQ_LEN, EMBED, SEQ_LEN, 1.0f);

  // Output projection
  wmma_gemm_kernel<<<gproj, blk, 0, stream>>>(attn, EMBED, Wo, EMBED, bo,
                                              out, nullptr, EMBED,
                                              ROWS, EMBED, EMBED, 1.0f);
}
